// MultiHeadAttention_58471684767876
// MI455X (gfx1250) — matrix operand order repaired and checked by reading, not compiled, not run on hardware
//
#include <hip/hip_runtime.h>
#include <hip/hip_bf16.h>

// ---------------------------------------------------------------------------
// MHA forward for MI455X (gfx1250): f16 WMMA (V_WMMA_F32_16X16X32_F16),
// f32 accumulate, flash-attention softmax, async global->LDS double buffering.
// ---------------------------------------------------------------------------

typedef _Float16 h4   __attribute__((ext_vector_type(4)));
typedef _Float16 h8   __attribute__((ext_vector_type(8)));
typedef _Float16 v16h __attribute__((ext_vector_type(16)));
typedef float    v8f  __attribute__((ext_vector_type(8)));
typedef int      i4   __attribute__((vector_size(16)));

#define HID 2048
#define NB  2
#define NS  2048
#define NH  16
#define HD  128

#if defined(__HIP_DEVICE_COMPILE__) && \
    __has_builtin(__builtin_amdgcn_global_load_async_to_lds_b128) && \
    __has_builtin(__builtin_amdgcn_s_wait_asynccnt)
#define USE_ASYNC_LDS 1
#else
#define USE_ASYNC_LDS 0
#endif

// Copy 16 bytes global -> LDS (async DMA if available, else through VGPRs).
__device__ __forceinline__ void cp16(_Float16* lds, const _Float16* g) {
#if USE_ASYNC_LDS
  __builtin_amdgcn_global_load_async_to_lds_b128(
      (i4 __attribute__((address_space(1)))*)(uintptr_t)g,
      (i4 __attribute__((address_space(3)))*)lds, 0, 0);
#else
  *(h8*)lds = *(const h8*)g;
#endif
}

__device__ __forceinline__ void wait_stage() {
#if USE_ASYNC_LDS
  __builtin_amdgcn_s_wait_asynccnt(0);
#endif
}

__device__ __forceinline__ v8f zero8() {
  v8f z;
#pragma unroll
  for (int i = 0; i < 8; ++i) z[i] = 0.0f;
  return z;
}

// Build a 16-half A-fragment from two 16-byte chunks (VGPR0-3 / VGPR4-7).
__device__ __forceinline__ v16h mk_frag(const _Float16* lo, const _Float16* hi) {
  union { v16h v; h8 h[2]; } u;
  u.h[0] = *(const h8*)lo;
  u.h[1] = *(const h8*)hi;
  return u.v;
}

__device__ __forceinline__ v8f wmma16(v16h a, v16h b, v8f c) {
  // D = A(16x32 f16) * B(32x16 f16) + C(16x16 f32)
  return __builtin_amdgcn_wmma_f32_16x16x32_f16(
      /*neg_a=*/false, a, /*neg_b=*/false, b,
      /*c_mod=*/(short)0, c, /*reuse_a=*/false, /*reuse_b=*/false);
}

// ---------------------------------------------------------------------------
// f32 -> f16 conversion (vectorized x4)
// ---------------------------------------------------------------------------
__global__ void cvt_f32_f16(const float* __restrict__ src,
                            _Float16* __restrict__ dst, int n) {
  int i = (blockIdx.x * blockDim.x + threadIdx.x) * 4;
  if (i < n) {
    float4 f = *(const float4*)(src + i);
    h4 o = { (_Float16)f.x, (_Float16)f.y, (_Float16)f.z, (_Float16)f.w };
    *(h4*)(dst + i) = o;
  }
}

// ---------------------------------------------------------------------------
// GEMM: Y[M,N] = A[M,K] * W[N,K]^T + bias   (M=4096, N=K=2048)
// mode 0: f16 row-major out; mode 1: f16 per-head-transposed out [B,H,D,S];
// mode 2: f32 row-major out.
// Block tile 128x128 (8 waves, 2x4), wave tile 64x32, k-step 32,
// double-buffered LDS with async global->LDS staging.
// ---------------------------------------------------------------------------
__global__ __launch_bounds__(256)
void gemm_xwT(const _Float16* __restrict__ A, const _Float16* __restrict__ W,
              const float* __restrict__ bias, _Float16* __restrict__ outH,
              float* __restrict__ outF, int mode) {
  alignas(16) __shared__ _Float16 sA[2][128 * 32];
  alignas(16) __shared__ _Float16 sB[2][128 * 32];

  const int t    = threadIdx.x;
  const int w    = t >> 5;
  const int lane = t & 31;
  const int l16  = lane & 15;
  const int hl   = lane >> 4;
  const int wm   = w >> 2;  // 2 row-groups of 64
  const int wn   = w & 3;   // 4 col-groups of 32
  const int m0   = blockIdx.y * 128;
  const int n0   = blockIdx.x * 128;

  v8f acc[4][2];
#pragma unroll
  for (int mi = 0; mi < 4; ++mi)
#pragma unroll
    for (int nj = 0; nj < 2; ++nj) acc[mi][nj] = zero8();

  auto stage = [&](int kt, int bufi) {
    const int k0 = kt * 32;
#pragma unroll
    for (int r = 0; r < 2; ++r) {
      int idx = t + r * 256;
      int row = idx >> 2, kc = (idx & 3) * 8;
      cp16(&sA[bufi][row * 32 + kc], &A[(m0 + row) * HID + k0 + kc]);
    }
#pragma unroll
    for (int r = 0; r < 2; ++r) {
      int idx = t + r * 256;
      int n = idx >> 2, kc = (idx & 3) * 8;
      cp16(&sB[bufi][n * 32 + kc], &W[(n0 + n) * HID + k0 + kc]);
    }
  };

  stage(0, 0);
  int buf = 0;

  for (int kt = 0; kt < HID / 32; ++kt) {
    wait_stage();
    __syncthreads();
    if (kt + 1 < HID / 32) stage(kt + 1, buf ^ 1);

    v16h a[4], b[2];
#pragma unroll
    for (int mi = 0; mi < 4; ++mi) {
      const int r = (wm * 64 + mi * 16 + l16) * 32;
      a[mi] = mk_frag(&sA[buf][r + hl * 8], &sA[buf][r + 16 + hl * 8]);
    }
#pragma unroll
    for (int nj = 0; nj < 2; ++nj) {
      const int r = (wn * 32 + nj * 16 + l16) * 32;
      b[nj] = mk_frag(&sB[buf][r + hl * 8], &sB[buf][r + 16 + hl * 8]);
    }
#pragma unroll
    for (int mi = 0; mi < 4; ++mi)
#pragma unroll
      for (int nj = 0; nj < 2; ++nj)
        acc[mi][nj] = wmma16(a[mi], b[nj], acc[mi][nj]);

    buf ^= 1;
  }

  // Epilogue: C VGPR j holds row (j + 8*half), col = lane%16 within subtile.
#pragma unroll
  for (int mi = 0; mi < 4; ++mi) {
#pragma unroll
    for (int nj = 0; nj < 2; ++nj) {
      const int col = n0 + wn * 32 + nj * 16 + l16;
      const float bv = bias[col];
#pragma unroll
      for (int jj = 0; jj < 8; ++jj) {
        const int row = m0 + wm * 64 + mi * 16 + jj + 8 * hl;
        const float v = acc[mi][nj][jj] + bv;
        if (mode == 0) {
          outH[row * HID + col] = (_Float16)v;
        } else if (mode == 2) {
          outF[row * HID + col] = v;
        } else {  // mode 1: V stored [B, H, D, S]
          const int b = row >> 11, s = row & (NS - 1);
          const int h = col >> 7, d = col & (HD - 1);
          outH[(((b * NH + h) * HD + d) << 11) + s] = (_Float16)v;
        }
      }
    }
  }
}

// ---------------------------------------------------------------------------
// Flash attention: grid (S/64, H, B), 128 threads = 4 waves.
// Wave w owns 16 query rows; 32-key tiles, double-buffered async K/V staging.
// Q/K are [B,S,HID] f16 (head h at cols h*128); Vt is [B,H,D,S] f16.
// ---------------------------------------------------------------------------
__global__ __launch_bounds__(128)
void flash_attn(const _Float16* __restrict__ Q, const _Float16* __restrict__ K,
                const _Float16* __restrict__ Vt, const float* __restrict__ mask,
                _Float16* __restrict__ ctx) {
  alignas(16) __shared__ _Float16 sK[2][32 * 128];   // [key][d]
  alignas(16) __shared__ _Float16 sV[2][128 * 32];   // [d][key]
  alignas(16) __shared__ _Float16 sP[4][16 * 32];    // per-wave P scratch

  const int t    = threadIdx.x;
  const int w    = t >> 5;
  const int lane = t & 31;
  const int l16  = lane & 15;
  const int hl   = lane >> 4;
  const int q0   = blockIdx.x * 64;
  const int h    = blockIdx.y;
  const int b    = blockIdx.z;

  auto stage_kv = [&](int kv0, int bufi) {
#pragma unroll
    for (int r = 0; r < 4; ++r) {  // K tile: 32 keys x 128 d
      int idx = t + r * 128;
      int row = idx >> 4, kc = (idx & 15) * 8;
      cp16(&sK[bufi][row * 128 + kc],
           &K[(size_t)(b * NS + kv0 + row) * HID + h * HD + kc]);
    }
#pragma unroll
    for (int r = 0; r < 4; ++r) {  // V^T tile: 128 d x 32 keys
      int idx = t + r * 128;
      int d = idx >> 2, kc = (idx & 3) * 8;
      cp16(&sV[bufi][d * 32 + kc],
           &Vt[((size_t)(b * NH + h) * HD + d) * NS + kv0 + kc]);
    }
  };

  // Resident Q fragments: 16 rows x 128 d = 4 A-frags (k=32 each)
  v16h qa[4];
  const _Float16* qbase = Q + (size_t)(b * NS + q0 + w * 16 + l16) * HID + h * HD;
#pragma unroll
  for (int kk = 0; kk < 4; ++kk)
    qa[kk] = mk_frag(qbase + kk * 32 + hl * 8, qbase + kk * 32 + 16 + hl * 8);

  float mrow[8], lrow[8];
  v8f o[8];
#pragma unroll
  for (int jj = 0; jj < 8; ++jj) { mrow[jj] = -3.0e30f; lrow[jj] = 0.0f; }
#pragma unroll
  for (int n = 0; n < 8; ++n) o[n] = zero8();

  const float scale = 0.0883883476483184f;  // 1/sqrt(128)

  stage_kv(0, 0);
  int buf = 0;

  for (int kv0 = 0; kv0 < NS; kv0 += 32) {
    wait_stage();
    __syncthreads();
    if (kv0 + 32 < NS) stage_kv(kv0 + 32, buf ^ 1);

    float mk[2];
#pragma unroll
    for (int sub = 0; sub < 2; ++sub)
      mk[sub] = mask[b * NS + kv0 + sub * 16 + l16];

    // Scores: S = Q * K^T  (2 key-subtiles x 4 k-steps)
    v8f s[2];
#pragma unroll
    for (int sub = 0; sub < 2; ++sub) {
      s[sub] = zero8();
#pragma unroll
      for (int ks = 0; ks < 4; ++ks) {
        v16h kb = mk_frag(&sK[buf][(sub * 16 + l16) * 128 + ks * 32 + hl * 8], &sK[buf][(sub * 16 + l16) * 128 + ks * 32 + 16 + hl * 8]);
        s[sub] = wmma16(qa[ks], kb, s[sub]);
      }
    }

    // Online softmax (rows live in 16-lane halves; xor 1/2/4/8 reduces a row)
    float p[2][8], rmax[8];
#pragma unroll
    for (int jj = 0; jj < 8; ++jj) {
      float v0 = s[0][jj] * scale + mk[0];
      float v1 = s[1][jj] * scale + mk[1];
      p[0][jj] = v0; p[1][jj] = v1;
      float r = fmaxf(v0, v1);
      r = fmaxf(r, __shfl_xor(r, 1));
      r = fmaxf(r, __shfl_xor(r, 2));
      r = fmaxf(r, __shfl_xor(r, 4));
      r = fmaxf(r, __shfl_xor(r, 8));
      rmax[jj] = r;
    }
#pragma unroll
    for (int jj = 0; jj < 8; ++jj) {
      float mnew = fmaxf(mrow[jj], rmax[jj]);
      float corr = __expf(mrow[jj] - mnew);
      mrow[jj] = mnew;
      float e0 = __expf(p[0][jj] - mnew);
      float e1 = __expf(p[1][jj] - mnew);
      p[0][jj] = e0; p[1][jj] = e1;
      float sum = e0 + e1;
      sum += __shfl_xor(sum, 1);
      sum += __shfl_xor(sum, 2);
      sum += __shfl_xor(sum, 4);
      sum += __shfl_xor(sum, 8);
      lrow[jj] = lrow[jj] * corr + sum;
#pragma unroll
      for (int n = 0; n < 8; ++n) o[n][jj] *= corr;
    }

    // P: C-layout -> A-layout via per-wave LDS round trip
#pragma unroll
    for (int sub = 0; sub < 2; ++sub)
#pragma unroll
      for (int jj = 0; jj < 8; ++jj)
        sP[w][(jj + 8 * hl) * 32 + sub * 16 + l16] = (_Float16)p[sub][jj];
    __syncthreads();  // hard-orders the LDS round trip

    // O += P * V   (one k-step over 32 keys, 8 d-subtiles)
    v16h pa = mk_frag(&sP[w][l16 * 32 + hl * 8],
                      &sP[w][l16 * 32 + 16 + hl * 8]);
#pragma unroll
    for (int n = 0; n < 8; ++n) {
      v16h vb = mk_frag(&sV[buf][(n * 16 + l16) * 32 + hl * 8], &sV[buf][(n * 16 + l16) * 32 + 16 + hl * 8]);
      o[n] = wmma16(pa, vb, o[n]);
    }

    buf ^= 1;
  }

  // Normalize and write ctx [B,S,HID] f16
#pragma unroll
  for (int jj = 0; jj < 8; ++jj) {
    const float inv = 1.0f / lrow[jj];
    const int srow = q0 + w * 16 + jj + 8 * hl;
#pragma unroll
    for (int n = 0; n < 8; ++n) {
      const int col = h * HD + n * 16 + l16;
      ctx[(size_t)(b * NS + srow) * HID + col] = (_Float16)(o[n][jj] * inv);
    }
  }
}

// ---------------------------------------------------------------------------
// Host-side orchestration
// ---------------------------------------------------------------------------
extern "C" void kernel_launch(void* const* d_in, const int* in_sizes, int n_in,
                              void* d_out, int out_size, void* d_ws, size_t ws_size,
                              hipStream_t stream) {
  const float* x    = (const float*)d_in[0];
  const float* mask = (const float*)d_in[1];
  const float* Wq   = (const float*)d_in[2];
  const float* bq   = (const float*)d_in[3];
  const float* Wk   = (const float*)d_in[4];
  const float* bk   = (const float*)d_in[5];
  const float* Wv   = (const float*)d_in[6];
  const float* bv   = (const float*)d_in[7];
  const float* Wo   = (const float*)d_in[8];
  const float* bo   = (const float*)d_in[9];
  float* out = (float*)d_out;

  const size_t SZX = (size_t)NB * NS * HID;  // 8388608
  const size_t SZW = (size_t)HID * HID;      // 4194304

  _Float16* ws  = (_Float16*)d_ws;
  _Float16* xh  = ws;
  _Float16* wqh = xh + SZX;
  _Float16* wkh = wqh + SZW;
  _Float16* wvh = wkh + SZW;
  _Float16* woh = wvh + SZW;
  _Float16* qh  = woh + SZW;
  _Float16* kh  = qh + SZX;
  _Float16* vth = kh + SZX;   // [B,H,D,S]
  _Float16* ch  = vth + SZX;

  cvt_f32_f16<<<(int)(SZX / 1024), 256, 0, stream>>>(x, xh, (int)SZX);
  cvt_f32_f16<<<(int)(SZW / 1024), 256, 0, stream>>>(Wq, wqh, (int)SZW);
  cvt_f32_f16<<<(int)(SZW / 1024), 256, 0, stream>>>(Wk, wkh, (int)SZW);
  cvt_f32_f16<<<(int)(SZW / 1024), 256, 0, stream>>>(Wv, wvh, (int)SZW);
  cvt_f32_f16<<<(int)(SZW / 1024), 256, 0, stream>>>(Wo, woh, (int)SZW);

  dim3 gg(HID / 128, (NB * NS) / 128);  // (16, 32)
  gemm_xwT<<<gg, 256, 0, stream>>>(xh, wqh, bq, qh, nullptr, 0);
  gemm_xwT<<<gg, 256, 0, stream>>>(xh, wkh, bk, kh, nullptr, 0);
  gemm_xwT<<<gg, 256, 0, stream>>>(xh, wvh, bv, vth, nullptr, 1);

  flash_attn<<<dim3(NS / 64, NH, NB), 128, 0, stream>>>(qh, kh, vth, mask, ch);

  gemm_xwT<<<gg, 256, 0, stream>>>(ch, woh, bo, nullptr, out, 2);
}
